// Decoder_89558658056844
// MI455X (gfx1250) — matrix operand order repaired and checked by reading, not compiled, not run on hardware
//
#include <hip/hip_runtime.h>
#include <math.h>

// ---------------------------------------------------------------------------
// Tacotron2-style decoder for MI455X (gfx1250).
// Design: single persistent kernel, 200 sequential steps, grid barriers via
// agent-scope atomics. All GEMMs are M=16 (batch) so every output tile is one
// v_wmma_f32_16x16x32_bf16 16x16 tile; weights are converted once per launch
// to bf16 in a K-major [Ntile][K][16] tiled layout (~81 MB -> L2 resident).
// ---------------------------------------------------------------------------

typedef unsigned short u16;
typedef __attribute__((ext_vector_type(16))) __bf16 v16bf;
typedef __attribute__((ext_vector_type(8)))  float  v8f;

union Frag { v16bf v; unsigned u[8]; };

#define AS        5632      // activation row stride (bf16 elems), per batch row
#define OFS_PA    0         // prenet-a output      (256)
#define OFS_CTX   256       // enc context          (512)
#define OFS_PB    768       // prenet-b output      (256)
#define OFS_CTXB  1024      // bert context         (512)
#define OFS_AH    1536      // attention-LSTM h     (1024)
#define OFS_AHB   2560      // attention-LSTM-b h   (1024)
#define OFS_DH    3584      // decoder-LSTM h       (1024)
#define OFS_DHB   4608      // decoder-LSTM-b h     (1024)

#define NWG       32
#define NTH       (NWG * 256)
#define T_OUT     200
#define GATE_OFS    256000  // 16*80*200
#define ALIGN_OFS   259200  // + 16*200
#define ALIGNB_OFS  796800  // + 16*200*168

struct Params {
  // raw f32 inputs
  const float *memory, *emb, *dec;
  const float *arnn_bias, *arnnb_bias, *drnn_bias, *drnnb_bias;
  const float *attn_conv, *attn_loc, *attn_v;
  const float *attnb_conv, *attnb_loc, *attnb_v;
  const float *proj_bias, *gate_wv, *gate_bias;
  // tiled bf16 weights (workspace)
  const u16 *wt_arnn, *wt_arnnb, *wt_drnn, *wt_drnnb;
  const u16 *wt_q, *wt_qb, *wt_p1a, *wt_p2a, *wt_p1b, *wt_p2b, *wt_proj;
  // state (workspace)
  u16*   act;
  float *gatesA, *gatesAB, *gatesD, *gatesDB;
  float *cA, *cAB, *cD, *cDB;
  float *aw, *awc, *awb, *awcb;
  float *fmap, *fmapb, *e, *eb, *pq, *pqb;
  const float *pmem, *pbert;
  unsigned* bar;
  float* out;
};

// ---------------- small helpers ----------------

__device__ __forceinline__ float b2f(u16 h) {
  unsigned u = ((unsigned)h) << 16;
  return __builtin_bit_cast(float, u);
}
__device__ __forceinline__ u16 f2b(float f) {
  unsigned u = __builtin_bit_cast(unsigned, f);
  unsigned r = u + 0x7FFFu + ((u >> 16) & 1u);   // round-to-nearest-even
  return (u16)(r >> 16);
}
__device__ __forceinline__ float sigm(float x) { return 1.f / (1.f + expf(-x)); }

__device__ void grid_sync(unsigned* bar) {
  __syncthreads();
  if (threadIdx.x == 0) {
    unsigned gen = __hip_atomic_load(&bar[1], __ATOMIC_RELAXED, __HIP_MEMORY_SCOPE_AGENT);
    unsigned arr = __hip_atomic_fetch_add(&bar[0], 1u, __ATOMIC_ACQ_REL, __HIP_MEMORY_SCOPE_AGENT);
    if (arr == NWG - 1) {
      __hip_atomic_store(&bar[0], 0u, __ATOMIC_RELAXED, __HIP_MEMORY_SCOPE_AGENT);
      __hip_atomic_fetch_add(&bar[1], 1u, __ATOMIC_RELEASE, __HIP_MEMORY_SCOPE_AGENT);
    } else {
      while (__hip_atomic_load(&bar[1], __ATOMIC_ACQUIRE, __HIP_MEMORY_SCOPE_AGENT) == gen)
        __builtin_amdgcn_s_sleep(1);
    }
  }
  __syncthreads();
}

// ---------------- WMMA 16x16 tile GEMM ----------------
// C[16, n0:n0+16] += sum over K segments of A[16, seg] * Wtile, bf16 in, f32 out.
// A: row-major bf16, row stride lda. Wtile: K-major [Ktot][16] bf16.
// A frag: lanes 0-15 hold M=0..15 (K pairs per VGPR per ISA layout);
// B frag: lanes hold N = lane%16, VGPR p holds the same K pair as A.
__device__ v8f wmma_tile(const u16* __restrict__ A, int lda,
                         const int* so, const int* sl, int ns,
                         const u16* __restrict__ wt)
{
  v8f acc = {0.f, 0.f, 0.f, 0.f, 0.f, 0.f, 0.f, 0.f};
  const int lane = threadIdx.x & 31;
  const int hf = lane >> 4;
  const int m  = lane & 15;
  int kw = 0;
  for (int s = 0; s < ns; ++s) {
    const u16* arow = A + (size_t)m * lda + so[s];
    const int klen = sl[s];
    for (int kb = 0; kb < klen; kb += 32) {
      Frag fa, fb;
      const u16* wrow = wt + (size_t)(kw + kb) * 16 + m;
      __builtin_prefetch((const void*)(wrow + (size_t)64 * 16), 0, 0);
#pragma unroll
      for (int p = 0; p < 8; ++p) {
        const int kk = ((p & 4) << 2) | ((p & 3) << 1) | (hf << 3);
        fa.u[p] = *(const unsigned*)(arow + kb + kk);
        fb.u[p] = (unsigned)wrow[(size_t)kk * 16] | ((unsigned)wrow[(size_t)(kk + 1) * 16] << 16);
      }
      acc = __builtin_amdgcn_wmma_f32_16x16x32_bf16(false, fa.v, false, fb.v,
                                                    (short)0, acc, false, false);
    }
    kw += klen;
  }
  return acc;
}

__device__ __forceinline__ void store_tile(float* dst, int ldd, int n0, v8f acc) {
  const int lane = threadIdx.x & 31;
  const int hf = lane >> 4, col = lane & 15;
#pragma unroll
  for (int r = 0; r < 8; ++r)
    dst[(size_t)(r + hf * 8) * ldd + n0 + col] = acc[r];
}

// ---------------- per-stage device code ----------------

// two-layer prenet for one side, executed by a whole workgroup (LDS staging)
__device__ void prenet_wg(const Params& P, int t, const u16* wt1, const u16* wt2,
                          int ofs_p, unsigned char* smem)
{
  u16* xl = (u16*)smem;              // [16][96] bf16, zero-padded K 80->96
  u16* p1 = (u16*)smem + 16 * 96;    // [16][256] bf16
  const int tid = threadIdx.x;
  for (int i = tid; i < 16 * 96; i += 256) {
    int b = i / 96, mch = i - b * 96;
    float v = 0.f;
    if (t > 0 && mch < 80) v = P.dec[((size_t)b * 80 + mch) * T_OUT + (t - 1)];
    xl[i] = f2b(v);
  }
  __syncthreads();
  const int wv = tid >> 5, lane = tid & 31, hf = lane >> 4, col = lane & 15;
  const int so0[1] = {0};
  const int sl96[1] = {96};
  const int sl256[1] = {256};
#pragma unroll
  for (int ti = 0; ti < 2; ++ti) {
    int tile = wv + ti * 8;
    v8f acc = wmma_tile(xl, 96, so0, sl96, 1, wt1 + (size_t)tile * 96 * 16);
#pragma unroll
    for (int r = 0; r < 8; ++r)
      p1[(size_t)(r + hf * 8) * 256 + tile * 16 + col] = f2b(fmaxf(acc[r], 0.f));
  }
  __syncthreads();
#pragma unroll
  for (int ti = 0; ti < 2; ++ti) {
    int tile = wv + ti * 8;
    v8f acc = wmma_tile(p1, 256, so0, sl256, 1, wt2 + (size_t)tile * 256 * 16);
#pragma unroll
    for (int r = 0; r < 8; ++r)
      P.act[(size_t)(r + hf * 8) * AS + ofs_p + tile * 16 + col] = f2b(fmaxf(acc[r], 0.f));
  }
  __syncthreads();
}

// location conv: fmap[b,f,t] = sum_{c,k} cat[b,c,t+k-15]*convW[f,c,k]
__device__ void fmap_stage(const Params& P, int i0, int step)
{
  const int NEf = 16 * 32 * 168, NBf = 16 * 32 * 96;
  for (int i = i0; i < NEf + NBf; i += step) {
    const bool enc = i < NEf;
    const int idx = enc ? i : i - NEf;
    const int T = enc ? 168 : 96;
    const int b = idx / (32 * T);
    const int rem = idx - b * 32 * T;
    const int f = rem / T, tt = rem - f * T;
    const float* cw   = enc ? P.attn_conv : P.attnb_conv;
    const float* awv  = (enc ? P.aw : P.awb) + b * T;
    const float* awcv = (enc ? P.awc : P.awcb) + b * T;
    const float* w0 = cw + (size_t)f * 62;
    const float* w1 = w0 + 31;
    float s = 0.f;
    for (int k = 0; k < 31; ++k) {
      int tp = tt + k - 15;
      if (tp >= 0 && tp < T) s += awv[tp] * w0[k] + awcv[tp] * w1[k];
    }
    (enc ? P.fmap : P.fmapb)[((size_t)b * 32 + f) * T + tt] = s;
  }
}

// e[b,t] = sum_a v[a]*tanh(pq[b,a] + ploc[b,t,a] + pmem[b,t,a])
__device__ void e_stage(const Params& P, int gt0)
{
  const int NEe = 16 * 168, NBe = 16 * 96;
  for (int i = gt0; i < NEe + NBe; i += NTH) {
    const bool enc = i < NEe;
    const int idx = enc ? i : i - NEe;
    const int T = enc ? 168 : 96;
    const int b = idx / T, tt = idx - b * T;
    const float* fm  = (enc ? P.fmap : P.fmapb) + (size_t)b * 32 * T + tt;
    const float* pqv = (enc ? P.pq : P.pqb) + b * 128;
    const float* pm  = (enc ? P.pmem : P.pbert) + ((size_t)b * T + tt) * 128;
    const float* lw  = enc ? P.attn_loc : P.attnb_loc;
    const float* vv  = enc ? P.attn_v : P.attnb_v;
    float fr[32];
#pragma unroll
    for (int f = 0; f < 32; ++f) fr[f] = fm[(size_t)f * T];
    float s = 0.f;
    for (int a = 0; a < 128; ++a) {
      float x = pqv[a] + pm[a];
      const float* lrow = lw + a * 32;
#pragma unroll
      for (int f = 0; f < 32; ++f) x += fr[f] * lrow[f];
      s += vv[a] * tanhf(x);
    }
    (enc ? P.e : P.eb)[b * T + tt] = s;
  }
}

// softmax over T, context, aw/awc update, alignment output; one wg per (b, side)
__device__ void attn_finish(const Params& P, int b, bool enc, int t, unsigned char* smem)
{
  const int T = enc ? 168 : 96;
  float* ev   = (enc ? P.e : P.eb) + b * T;
  float* awv  = (enc ? P.aw : P.awb) + b * T;
  float* awcv = (enc ? P.awc : P.awcb) + b * T;
  const float* mem = (enc ? P.memory : P.emb) + (size_t)b * T * 512;
  const int cofs = enc ? OFS_CTX : OFS_CTXB;
  float* oal = P.out + (enc ? ALIGN_OFS : ALIGNB_OFS) + ((size_t)b * T_OUT + t) * T;
  float* red  = (float*)smem;        // [256]
  float* wbuf = red + 256;           // [T]
  const int tid = threadIdx.x;

  float mx = -1e30f;
  for (int i = tid; i < T; i += 256) mx = fmaxf(mx, ev[i]);
  red[tid] = mx; __syncthreads();
  for (int s = 128; s > 0; s >>= 1) { if (tid < s) red[tid] = fmaxf(red[tid], red[tid + s]); __syncthreads(); }
  mx = red[0]; __syncthreads();

  float sum = 0.f;
  for (int i = tid; i < T; i += 256) { float v = expf(ev[i] - mx); wbuf[i] = v; sum += v; }
  red[tid] = sum; __syncthreads();
  for (int s = 128; s > 0; s >>= 1) { if (tid < s) red[tid] += red[tid + s]; __syncthreads(); }
  float inv = 1.f / red[0]; __syncthreads();

  for (int i = tid; i < T; i += 256) {
    float w = wbuf[i] * inv;
    wbuf[i] = w; awv[i] = w; awcv[i] += w; oal[i] = w;
  }
  __syncthreads();

  for (int n = tid; n < 512; n += 256) {
    float s = 0.f;
    for (int tt = 0; tt < T; ++tt) s += wbuf[tt] * mem[(size_t)tt * 512 + n];
    P.act[(size_t)b * AS + cofs + n] = f2b(s);
  }
  __syncthreads();
}

// ---------------- persistent decoder kernel ----------------

__global__ __launch_bounds__(256) void k_decoder(Params P)
{
  __shared__ __align__(16) unsigned char smem[12544];
  const int wg = blockIdx.x;
  const int tid = threadIdx.x;
  const int lane = tid & 31;
  const int wv = tid >> 5;
  const int gw = wg * 8 + wv;     // global wave id: 0..255
  const int gt = wg * 256 + tid;  // global thread id: 0..8191

  // stage 0: prenet(0) + fmap(0)
  if (wg == 0)      prenet_wg(P, 0, P.wt_p1a, P.wt_p2a, OFS_PA, smem);
  else if (wg == 1) prenet_wg(P, 0, P.wt_p1b, P.wt_p2b, OFS_PB, smem);
  else              fmap_stage(P, gt - 512, NTH - 512);
  grid_sync(P.bar);

  for (int t = 0; t < T_OUT; ++t) {
    // S1: attention-LSTM gates (a and b), 512 tiles over 256 waves
    {
      const int soA[3] = {OFS_PA,  OFS_CTX,  OFS_AH};
      const int soB[3] = {OFS_PB,  OFS_CTXB, OFS_AHB};
      const int sl3[3] = {256, 512, 1024};
      for (int tt = gw; tt < 512; tt += 256) {
        if (tt < 256) {
          v8f acc = wmma_tile(P.act, AS, soA, sl3, 3, P.wt_arnn + (size_t)tt * 1792 * 16);
          store_tile(P.gatesA, 4096, tt * 16, acc);
        } else {
          v8f acc = wmma_tile(P.act, AS, soB, sl3, 3, P.wt_arnnb + (size_t)(tt - 256) * 1792 * 16);
          store_tile(P.gatesAB, 4096, (tt - 256) * 16, acc);
        }
      }
    }
    grid_sync(P.bar);

    // S2: pointwise LSTM update for a/ab
    for (int i = gt; i < 32768; i += NTH) {
      int which = i >> 14, r = i & 16383, b = r >> 10, j = r & 1023;
      const float* G  = which ? P.gatesAB : P.gatesA;
      const float* bs = which ? P.arnnb_bias : P.arnn_bias;
      float* C = which ? P.cAB : P.cA;
      int ho = which ? OFS_AHB : OFS_AH;
      float ig = sigm(G[b * 4096 + j] + bs[j]);
      float fg = sigm(G[b * 4096 + 1024 + j] + bs[1024 + j]);
      float gg = tanhf(G[b * 4096 + 2048 + j] + bs[2048 + j]);
      float og = sigm(G[b * 4096 + 3072 + j] + bs[3072 + j]);
      float c = fg * C[b * 1024 + j] + ig * gg;
      C[b * 1024 + j] = c;
      P.act[(size_t)b * AS + ho + j] = f2b(og * tanhf(c));
    }
    grid_sync(P.bar);

    // S3a: attention query projections (16 WMMA tiles)
    if (gw < 16) {
      const bool enc = gw < 8;
      const int tile = gw & 7;
      const int so1[1] = { enc ? OFS_AH : OFS_AHB };
      const int sl1[1] = { 1024 };
      v8f acc = wmma_tile(P.act, AS, so1, sl1, 1,
                          (enc ? P.wt_q : P.wt_qb) + (size_t)tile * 1024 * 16);
      store_tile(enc ? P.pq : P.pqb, 128, tile * 16, acc);
    }
    grid_sync(P.bar);

    // S3b: energies
    e_stage(P, gt);
    grid_sync(P.bar);

    // S3c: softmax + context + carry updates (wg 0..15 enc, 16..31 bert)
    if (wg < 16) attn_finish(P, wg, true, t, smem);
    else         attn_finish(P, wg - 16, false, t, smem);
    grid_sync(P.bar);

    // S4: decoder-LSTM gates (d and db)
    {
      const int soD[5]  = {OFS_AH, OFS_AHB, OFS_CTX, OFS_CTXB, OFS_DH};
      const int slD[5]  = {1024, 1024, 512, 512, 1024};
      const int soDB[3] = {OFS_AHB, OFS_CTXB, OFS_DHB};
      const int slDB[3] = {1024, 512, 1024};
      for (int tt = gw; tt < 512; tt += 256) {
        if (tt < 256) {
          v8f acc = wmma_tile(P.act, AS, soD, slD, 5, P.wt_drnn + (size_t)tt * 4096 * 16);
          store_tile(P.gatesD, 4096, tt * 16, acc);
        } else {
          v8f acc = wmma_tile(P.act, AS, soDB, slDB, 3, P.wt_drnnb + (size_t)(tt - 256) * 2560 * 16);
          store_tile(P.gatesDB, 4096, (tt - 256) * 16, acc);
        }
      }
    }
    grid_sync(P.bar);

    // S5: pointwise LSTM update for d/db
    for (int i = gt; i < 32768; i += NTH) {
      int which = i >> 14, r = i & 16383, b = r >> 10, j = r & 1023;
      const float* G  = which ? P.gatesDB : P.gatesD;
      const float* bs = which ? P.drnnb_bias : P.drnn_bias;
      float* C = which ? P.cDB : P.cD;
      int ho = which ? OFS_DHB : OFS_DH;
      float ig = sigm(G[b * 4096 + j] + bs[j]);
      float fg = sigm(G[b * 4096 + 1024 + j] + bs[1024 + j]);
      float gg = tanhf(G[b * 4096 + 2048 + j] + bs[2048 + j]);
      float og = sigm(G[b * 4096 + 3072 + j] + bs[3072 + j]);
      float c = fg * C[b * 1024 + j] + ig * gg;
      C[b * 1024 + j] = c;
      P.act[(size_t)b * AS + ho + j] = f2b(og * tanhf(c));
    }
    grid_sync(P.bar);

    // S6: projection + gate + prenet(t+1) + fmap(t+1) (independent work fused)
    if (wg == 0) {
      if (wv < 5) {
        const int soP[3] = {OFS_DH, OFS_CTX, OFS_CTXB};
        const int slP[3] = {1024, 512, 512};
        v8f acc = wmma_tile(P.act, AS, soP, slP, 3, P.wt_proj + (size_t)wv * 2048 * 16);
        const int hf = lane >> 4, col = lane & 15;
#pragma unroll
        for (int r = 0; r < 8; ++r) {
          int b = r + hf * 8, n = wv * 16 + col;
          P.out[((size_t)b * 80 + n) * T_OUT + t] = acc[r] + P.proj_bias[n];
        }
      } else if (wv == 5 && lane < 16) {
        int b = lane;
        float s = P.gate_bias[0];
        const u16* h = P.act + (size_t)b * AS;
        for (int k = 0; k < 1024; ++k) s += b2f(h[OFS_DH + k])   * P.gate_wv[k];
        for (int k = 0; k < 512;  ++k) s += b2f(h[OFS_CTX + k])  * P.gate_wv[1024 + k];
        for (int k = 0; k < 512;  ++k) s += b2f(h[OFS_CTXB + k]) * P.gate_wv[1536 + k];
        P.out[GATE_OFS + (size_t)b * T_OUT + t] = s;
      }
    } else if (wg == 1) { if (t + 1 < T_OUT) prenet_wg(P, t + 1, P.wt_p1a, P.wt_p2a, OFS_PA, smem); }
    else if (wg == 2)   { if (t + 1 < T_OUT) prenet_wg(P, t + 1, P.wt_p1b, P.wt_p2b, OFS_PB, smem); }
    else                { if (t + 1 < T_OUT) fmap_stage(P, gt - 768, NTH - 768); }
    grid_sync(P.bar);
  }
}

// ---------------- setup kernels ----------------

__global__ void k_zero(unsigned* p, size_t n)
{
  size_t stride = (size_t)gridDim.x * blockDim.x;
  for (size_t i = (size_t)blockIdx.x * blockDim.x + threadIdx.x; i < n; i += stride)
    p[i] = 0u;
}

// Convert f32 weights (wih rows [N][Kih] + optional whh [N][Khh]) into bf16
// K-major tiles: dst[tile][k][n], K zero-padded Kih->Kpih.
__global__ void k_convert(const float* __restrict__ ih, const float* __restrict__ hh,
                          int Kih, int Kpih, int Khh, int ntiles, u16* __restrict__ dst)
{
  const int Ktot = Kpih + Khh;
  const size_t total = (size_t)ntiles * Ktot * 16;
  size_t stride = (size_t)gridDim.x * blockDim.x;
  for (size_t i = (size_t)blockIdx.x * blockDim.x + threadIdx.x; i < total; i += stride) {
    int n = (int)(i % 16);
    size_t q = i / 16;
    int k = (int)(q % Ktot);
    int tile = (int)(q / Ktot);
    int row = tile * 16 + n;
    float v;
    if (k < Kpih) v = (k < Kih) ? ih[(size_t)row * Kih + k] : 0.f;
    else          v = hh[(size_t)row * Khh + (k - Kpih)];
    dst[i] = f2b(v);
  }
}

// pmem[b,t,a] = sum_e mem[b,t,e] * am[a,e]
__global__ void k_pmem(const float* __restrict__ mem, const float* __restrict__ am,
                       int T, float* __restrict__ dst)
{
  const size_t total = (size_t)16 * T * 128;
  size_t stride = (size_t)gridDim.x * blockDim.x;
  for (size_t i = (size_t)blockIdx.x * blockDim.x + threadIdx.x; i < total; i += stride) {
    int a = (int)(i % 128);
    size_t q = i / 128;
    int tt = (int)(q % T);
    int b = (int)(q / T);
    const float* mrow = mem + ((size_t)b * T + tt) * 512;
    const float* arow = am + (size_t)a * 512;
    float s = 0.f;
    for (int e = 0; e < 512; ++e) s += mrow[e] * arow[e];
    dst[i] = s;
  }
}

// ---------------- host launcher ----------------

extern "C" void kernel_launch(void* const* d_in, const int* in_sizes, int n_in,
                              void* d_out, int out_size, void* d_ws, size_t ws_size,
                              hipStream_t stream)
{
  (void)in_sizes; (void)n_in; (void)out_size; (void)ws_size;

  const float* memory_p   = (const float*)d_in[0];
  const float* emb_p      = (const float*)d_in[1];
  const float* dec_p      = (const float*)d_in[2];
  const float* pre_w1     = (const float*)d_in[3];
  const float* pre_w2     = (const float*)d_in[4];
  const float* preb_w1    = (const float*)d_in[5];
  const float* preb_w2    = (const float*)d_in[6];
  const float* arnn_wih   = (const float*)d_in[7];
  const float* arnn_whh   = (const float*)d_in[8];
  const float* arnn_b     = (const float*)d_in[9];
  const float* arnnb_wih  = (const float*)d_in[10];
  const float* arnnb_whh  = (const float*)d_in[11];
  const float* arnnb_b    = (const float*)d_in[12];
  const float* attn_q     = (const float*)d_in[13];
  const float* attn_m     = (const float*)d_in[14];
  const float* attn_conv  = (const float*)d_in[15];
  const float* attn_loc   = (const float*)d_in[16];
  const float* attn_v     = (const float*)d_in[17];
  const float* attnb_q    = (const float*)d_in[18];
  const float* attnb_m    = (const float*)d_in[19];
  const float* attnb_conv = (const float*)d_in[20];
  const float* attnb_loc  = (const float*)d_in[21];
  const float* attnb_v    = (const float*)d_in[22];
  const float* drnn_wih   = (const float*)d_in[23];
  const float* drnn_whh   = (const float*)d_in[24];
  const float* drnn_b     = (const float*)d_in[25];
  const float* drnnb_wih  = (const float*)d_in[26];
  const float* drnnb_whh  = (const float*)d_in[27];
  const float* drnnb_b    = (const float*)d_in[28];
  const float* proj_w     = (const float*)d_in[29];
  const float* proj_b     = (const float*)d_in[30];
  const float* gate_w     = (const float*)d_in[31];
  const float* gate_b     = (const float*)d_in[32];

  char* base = (char*)d_ws;
  size_t off = 0;
  auto alloc = [&](size_t bytes) -> void* {
    void* p = base + off;
    off = (off + bytes + 255) & ~(size_t)255;
    return p;
  };

  Params P;
  // ---- zero-initialized block (carries + barrier) ----
  P.act  = (u16*)  alloc((size_t)16 * AS * 2);
  P.cA   = (float*)alloc((size_t)16 * 1024 * 4);
  P.cAB  = (float*)alloc((size_t)16 * 1024 * 4);
  P.cD   = (float*)alloc((size_t)16 * 1024 * 4);
  P.cDB  = (float*)alloc((size_t)16 * 1024 * 4);
  P.aw   = (float*)alloc((size_t)16 * 168 * 4);
  P.awc  = (float*)alloc((size_t)16 * 168 * 4);
  P.awb  = (float*)alloc((size_t)16 * 96 * 4);
  P.awcb = (float*)alloc((size_t)16 * 96 * 4);
  P.bar  = (unsigned*)alloc(256);
  const size_t zero_bytes = off;
  // ---- write-before-read scratch ----
  P.gatesA  = (float*)alloc((size_t)16 * 4096 * 4);
  P.gatesAB = (float*)alloc((size_t)16 * 4096 * 4);
  P.gatesD  = (float*)alloc((size_t)16 * 4096 * 4);
  P.gatesDB = (float*)alloc((size_t)16 * 4096 * 4);
  P.pq    = (float*)alloc((size_t)16 * 128 * 4);
  P.pqb   = (float*)alloc((size_t)16 * 128 * 4);
  P.e     = (float*)alloc((size_t)16 * 168 * 4);
  P.eb    = (float*)alloc((size_t)16 * 96 * 4);
  P.fmap  = (float*)alloc((size_t)16 * 32 * 168 * 4);
  P.fmapb = (float*)alloc((size_t)16 * 32 * 96 * 4);
  float* pmem_w  = (float*)alloc((size_t)16 * 168 * 128 * 4);
  float* pbert_w = (float*)alloc((size_t)16 * 96 * 128 * 4);
  P.pmem = pmem_w; P.pbert = pbert_w;
  // ---- bf16 tiled weights ----
  u16* wt_arnn  = (u16*)alloc((size_t)256 * 1792 * 16 * 2);
  u16* wt_arnnb = (u16*)alloc((size_t)256 * 1792 * 16 * 2);
  u16* wt_drnn  = (u16*)alloc((size_t)256 * 4096 * 16 * 2);
  u16* wt_drnnb = (u16*)alloc((size_t)256 * 2560 * 16 * 2);
  u16* wt_q     = (u16*)alloc((size_t)8 * 1024 * 16 * 2);
  u16* wt_qb    = (u16*)alloc((size_t)8 * 1024 * 16 * 2);
  u16* wt_p1a   = (u16*)alloc((size_t)16 * 96 * 16 * 2);
  u16* wt_p2a   = (u16*)alloc((size_t)16 * 256 * 16 * 2);
  u16* wt_p1b   = (u16*)alloc((size_t)16 * 96 * 16 * 2);
  u16* wt_p2b   = (u16*)alloc((size_t)16 * 256 * 16 * 2);
  u16* wt_proj  = (u16*)alloc((size_t)5 * 2048 * 16 * 2);
  P.wt_arnn = wt_arnn; P.wt_arnnb = wt_arnnb;
  P.wt_drnn = wt_drnn; P.wt_drnnb = wt_drnnb;
  P.wt_q = wt_q; P.wt_qb = wt_qb;
  P.wt_p1a = wt_p1a; P.wt_p2a = wt_p2a; P.wt_p1b = wt_p1b; P.wt_p2b = wt_p2b;
  P.wt_proj = wt_proj;

  P.memory = memory_p; P.emb = emb_p; P.dec = dec_p;
  P.arnn_bias = arnn_b; P.arnnb_bias = arnnb_b;
  P.drnn_bias = drnn_b; P.drnnb_bias = drnnb_b;
  P.attn_conv = attn_conv; P.attn_loc = attn_loc; P.attn_v = attn_v;
  P.attnb_conv = attnb_conv; P.attnb_loc = attnb_loc; P.attnb_v = attnb_v;
  P.proj_bias = proj_b; P.gate_wv = gate_w; P.gate_bias = gate_b;
  P.out = (float*)d_out;

  // 1) zero carries + barrier state (deterministic across graph replays)
  k_zero<<<64, 256, 0, stream>>>((unsigned*)d_ws, zero_bytes / 4);

  // 2) weight conversion to tiled bf16 (one pass per launch; ~85 MB)
  auto cvt = [&](const float* ih, const float* hh, int Kih, int Kpih, int Khh,
                 int ntiles, u16* dst) {
    size_t total = (size_t)ntiles * (Kpih + Khh) * 16;
    int blocks = (int)((total + 255) / 256);
    if (blocks > 4096) blocks = 4096;
    k_convert<<<blocks, 256, 0, stream>>>(ih, hh, Kih, Kpih, Khh, ntiles, dst);
  };
  cvt(arnn_wih,  arnn_whh,  768,  768,  1024, 256, wt_arnn);
  cvt(arnnb_wih, arnnb_whh, 768,  768,  1024, 256, wt_arnnb);
  cvt(drnn_wih,  drnn_whh,  3072, 3072, 1024, 256, wt_drnn);
  cvt(drnnb_wih, drnnb_whh, 1536, 1536, 1024, 256, wt_drnnb);
  cvt(attn_q,  nullptr, 1024, 1024, 0, 8,  wt_q);
  cvt(attnb_q, nullptr, 1024, 1024, 0, 8,  wt_qb);
  cvt(pre_w1,  nullptr, 80,   96,   0, 16, wt_p1a);
  cvt(pre_w2,  nullptr, 256,  256,  0, 16, wt_p2a);
  cvt(preb_w1, nullptr, 80,   96,   0, 16, wt_p1b);
  cvt(preb_w2, nullptr, 256,  256,  0, 16, wt_p2b);
  cvt(proj_w,  nullptr, 2048, 2048, 0, 5,  wt_proj);

  // 3) precompute projected memories (done once, not per step)
  k_pmem<<<1344, 256, 0, stream>>>(memory_p, attn_m,  168, pmem_w);
  k_pmem<<<768,  256, 0, stream>>>(emb_p,    attnb_m, 96,  pbert_w);

  // 4) persistent decoder (32 WGs, 8 waves each; grid barriers between stages)
  k_decoder<<<dim3(NWG), dim3(256), 0, stream>>>(P);
}
